// DistVisionTransformer_24747601560226
// MI455X (gfx1250) — compile-verified, matrix operand order checked by reading, not run on hardware
//
#include <hip/hip_runtime.h>
#include <hip/hip_bf16.h>

#define USE_TDM 1   // Tensor Data Mover staging (set 0 to revert to coop loads)

// ---------------------------------------------------------------------------
// Types
// ---------------------------------------------------------------------------
typedef _Float16 h16;
typedef h16      v16h __attribute__((ext_vector_type(16)));
typedef float    v8f  __attribute__((ext_vector_type(8)));
typedef unsigned v4u  __attribute__((ext_vector_type(4)));
typedef unsigned v8u  __attribute__((ext_vector_type(8)));

// ---------------------------------------------------------------------------
// Model dimensions
// ---------------------------------------------------------------------------
constexpr int Bsz    = 8;
constexpr int E      = 768;
constexpr int Hh     = 12;
constexpr int Lyr    = 12;
constexpr int NCLS   = 1000;
constexpr int Dh     = 64;
constexpr int MLPD   = 3072;
constexpr int NPATCH = 196;
constexpr int Ntok   = 197;          // 196 patches + cls
constexpr int NPAD   = 224;          // Ntok padded to multiple of 32 for WMMA K
constexpr int BN     = Bsz * Ntok;   // 1576 token rows
constexpr int BH     = Bsz * Hh;     // 96 attention batches
constexpr int PE_ROWS= Bsz * NPATCH; // 1568 patch rows
constexpr int QKV3E  = 3 * E;        // 2304
constexpr float SCALEQ = 0.125f;     // 64^-0.5

// ---------------------------------------------------------------------------
// WMMA GEMM:  C[M,N] = A[M,K](f16) * W[N,K](f16)^T  (+bias[N]) (+activation)
// Block tile 128x64, K-step 32, 8 wave32 waves each doing 32x32 (2x2 WMMA).
// Tiles staged by the Tensor Data Mover into double-buffered LDS; descriptor
// padding reproduces the 40-half row stride (bank-conflict-free frag reads),
// and tensor_dim OOB handling zero-fills partial tiles.
// act: 0 = none, 1 = exact GELU, 2 = elu(x)+1
// ---------------------------------------------------------------------------
#define TM 128
#define TN 64
#define TK 32
#define LDSS 40   // LDS row stride in halfs (80 B)

constexpr unsigned OFF_A0 = 0;
constexpr unsigned OFF_B0 = TM * LDSS * 2;                    // 10240
constexpr unsigned OFF_A1 = (TM + TN) * LDSS * 2;             // 15360
constexpr unsigned OFF_B1 = OFF_A1 + TM * LDSS * 2;           // 25600
constexpr int      SMEMB  = 2 * (TM + TN) * LDSS * (int)sizeof(h16);  // 30720

__device__ inline v16h frag_ld(const h16* p) {
  // ISA 16-bit A/B frag: vec[0..7] = K k0..k0+7, vec[8..15] = K k0+16..k0+23
  v16h v;
  reinterpret_cast<uint4*>(&v)[0] = reinterpret_cast<const uint4*>(p)[0];
  reinterpret_cast<uint4*>(&v)[1] = reinterpret_cast<const uint4*>(p + 16)[0];
  return v;
}

#if USE_TDM
// Issue one TDM 2D tile load: global [tileM x tileK] (row stride strideElems,
// 2-byte elements) -> LDS at byte offset ldsOff with 16B pad every 64B
// (=> LDSS*2 = 80B LDS row stride). tensK/tensM bound the tensor for OOB
// zero-fill of partial tiles. Wave-uniform; issue from one wave only.
__device__ inline void tdm_load_2d(const h16* gsrc, unsigned ldsOff,
                                   int tensK, int tensM,
                                   int tileK, int tileM, int strideElems)
{
  unsigned long long ga = (unsigned long long)(size_t)gsrc;
  v4u g0;
  g0[0] = 1u;                                        // count=1, user mode
  g0[1] = ldsOff;                                    // lds_addr (bytes)
  g0[2] = (unsigned)ga;                              // global_addr[31:0]
  g0[3] = (unsigned)((ga >> 32) & 0x01ffffffu)       // global_addr[56:32]
        | (2u << 30);                                // type = 2 ("image")
  v8u g1;
  g1[0] = (1u << 16)      // data_size = 2 bytes
        | (1u << 20)      // pad_enable
        | (3u << 22)      // pad_interval: every 16 DWORDs (64 B)
        | (3u << 25);     // pad_amount:   4 DWORDs (16 B)
  unsigned td0 = (unsigned)tensK, td1 = (unsigned)tensM;
  g1[1] = (td0 & 0xffffu) << 16;                     // tensor_dim0 lo16
  g1[2] = (td0 >> 16) | ((td1 & 0xffffu) << 16);     // tensor_dim0 hi | dim1 lo
  g1[3] = (td1 >> 16) | ((unsigned)tileK << 16);     // dim1 hi | tile_dim0
  g1[4] = (unsigned)tileM & 0xffffu;                 // tile_dim1 (tile_dim2=0)
  unsigned long long st = (unsigned long long)(unsigned)strideElems;
  g1[5] = (unsigned)st;                              // tensor_dim0_stride lo
  g1[6] = (unsigned)(st >> 32) & 0xffffu;            // stride hi (dim1 stride=0)
  g1[7] = 0u;
  asm volatile("tensor_load_to_lds %0, %1" : : "s"(g0), "s"(g1) : "memory");
}
#endif

__global__ __launch_bounds__(256) void wmma_gemm_f16(
    const h16* __restrict__ A, long long sA, int lda,
    const h16* __restrict__ W, long long sW, int ldw,
    float* __restrict__ C, long long sC, int ldc,
    const float* __restrict__ bias,
    h16* __restrict__ C16, int ldc16,
    int M, int N, int K, int act)
{
  extern __shared__ __align__(16) h16 smem[];

  const int t     = threadIdx.x;
  const int wv    = t >> 5;
  const int ln    = t & 31;
  const int mBase = blockIdx.y * TM;
  const int nBase = blockIdx.x * TN;
  const int bat   = blockIdx.z;

  const h16* Ab = A + (long long)bat * sA;
  const h16* Wb = W + (long long)bat * sW;
  float*     Cb = C + (long long)bat * sC;

  const int wm = (wv >> 1) * 32;   // wave M offset in tile: 0/32/64/96
  const int wn = (wv & 1) * 32;    // wave N offset in tile: 0/32
  const int fr = ln & 15;
  const int fk = (ln >> 4) * 8;

  v8f acc00 = {}, acc01 = {}, acc10 = {}, acc11 = {};

#if USE_TDM
  const int nk = (K + TK - 1) / TK;
  auto issue = [&](int ki, int buf) {
    int k0 = ki * TK;
    tdm_load_2d(Ab + (long long)mBase * lda + k0, buf ? OFF_A1 : OFF_A0,
                K - k0, M - mBase, TK, TM, lda);
    tdm_load_2d(Wb + (long long)nBase * ldw + k0, buf ? OFF_B1 : OFF_B0,
                K - k0, N - nBase, TK, TN, ldw);
  };
  if (t < 32) issue(0, 0);                  // prologue: stage 0 in flight
  for (int ki = 0; ki < nk; ++ki) {
    int buf = ki & 1;
    if (t < 32) {
      if (ki + 1 < nk) {                    // overlap: next tiles via TDM
        issue(ki + 1, 1 - buf);
        __builtin_amdgcn_s_wait_tensorcnt(2);   // stage ki complete
      } else {
        __builtin_amdgcn_s_wait_tensorcnt(0);
      }
    }
    __syncthreads();
    const h16* As = smem + (buf ? (TM + TN) * LDSS : 0);
    const h16* Bs = As + TM * LDSS;
    const h16* pa = &As[(wm + fr) * LDSS + fk];
    const h16* pb = &Bs[(wn + fr) * LDSS + fk];
    v16h a0 = frag_ld(pa);
    v16h a1 = frag_ld(pa + 16 * LDSS);
    v16h b0 = frag_ld(pb);
    v16h b1 = frag_ld(pb + 16 * LDSS);
    acc00 = __builtin_amdgcn_wmma_f32_16x16x32_f16(false, a0, false, b0, (short)0, acc00, false, false);
    acc01 = __builtin_amdgcn_wmma_f32_16x16x32_f16(false, a0, false, b1, (short)0, acc01, false, false);
    acc10 = __builtin_amdgcn_wmma_f32_16x16x32_f16(false, a1, false, b0, (short)0, acc10, false, false);
    acc11 = __builtin_amdgcn_wmma_f32_16x16x32_f16(false, a1, false, b1, (short)0, acc11, false, false);
    __syncthreads();
  }
#else
  h16* As = smem;
  h16* Bs = smem + TM * LDSS;
  for (int k0 = 0; k0 < K; k0 += TK) {
    for (int s = t; s < (TM * TK) / 8; s += 256) {
      int row = s >> 2, ks = (s & 3) << 3;
      int gm = mBase + row, gk = k0 + ks;
      h16* dst = &As[row * LDSS + ks];
      if (gm < M && gk + 8 <= K) {
        *reinterpret_cast<uint4*>(dst) =
            *reinterpret_cast<const uint4*>(&Ab[(long long)gm * lda + gk]);
      } else {
        for (int e = 0; e < 8; ++e)
          dst[e] = (gm < M && (gk + e) < K) ? Ab[(long long)gm * lda + gk + e]
                                            : (h16)0.f;
      }
    }
    for (int s = t; s < (TN * TK) / 8; s += 256) {
      int row = s >> 2, ks = (s & 3) << 3;
      int gn = nBase + row, gk = k0 + ks;
      h16* dst = &Bs[row * LDSS + ks];
      if (gn < N && gk + 8 <= K) {
        *reinterpret_cast<uint4*>(dst) =
            *reinterpret_cast<const uint4*>(&Wb[(long long)gn * ldw + gk]);
      } else {
        for (int e = 0; e < 8; ++e)
          dst[e] = (gn < N && (gk + e) < K) ? Wb[(long long)gn * ldw + gk + e]
                                            : (h16)0.f;
      }
    }
    __syncthreads();
    const h16* pa = &As[(wm + fr) * LDSS + fk];
    const h16* pb = &Bs[(wn + fr) * LDSS + fk];
    v16h a0 = frag_ld(pa);
    v16h a1 = frag_ld(pa + 16 * LDSS);
    v16h b0 = frag_ld(pb);
    v16h b1 = frag_ld(pb + 16 * LDSS);
    acc00 = __builtin_amdgcn_wmma_f32_16x16x32_f16(false, a0, false, b0, (short)0, acc00, false, false);
    acc01 = __builtin_amdgcn_wmma_f32_16x16x32_f16(false, a0, false, b1, (short)0, acc01, false, false);
    acc10 = __builtin_amdgcn_wmma_f32_16x16x32_f16(false, a1, false, b0, (short)0, acc10, false, false);
    acc11 = __builtin_amdgcn_wmma_f32_16x16x32_f16(false, a1, false, b1, (short)0, acc11, false, false);
    __syncthreads();
  }
#endif

  // ---- epilogue + store (C/D layout: VGPR r -> M=r+8*(ln>>4), N=ln&15) ----
  auto store_tile = [&](const v8f& acc, int mt, int nt) {
    int gn = nBase + wn + nt * 16 + fr;
    if (gn >= N) return;
    float bv = bias ? bias[gn] : 0.f;
    int gm0 = mBase + wm + mt * 16 + ((ln >> 4) << 3);
    for (int r = 0; r < 8; ++r) {
      int gm = gm0 + r;
      if (gm < M) {
        float v = acc[r] + bv;
        if (act == 1)      v = 0.5f * v * (1.f + erff(v * 0.70710678118f));
        else if (act == 2) v = (v > 0.f) ? v + 1.f : __expf(v);
        Cb[(long long)gm * ldc + gn] = v;
        if (C16) C16[(long long)gm * ldc16 + gn] = (h16)v;
      }
    }
  };
  store_tile(acc00, 0, 0);
  store_tile(acc01, 0, 1);
  store_tile(acc10, 1, 0);
  store_tile(acc11, 1, 1);
}

// ---------------------------------------------------------------------------
// Reductions (wave32)
// ---------------------------------------------------------------------------
__device__ inline float wave_sum(float v) {
  for (int m = 16; m >= 1; m >>= 1) v += __shfl_xor(v, m, 32);
  return v;
}
__device__ inline float wave_max(float v) {
  for (int m = 16; m >= 1; m >>= 1) v = fmaxf(v, __shfl_xor(v, m, 32));
  return v;
}
__device__ inline float block_sum(float v, float* sh) {
  int t = threadIdx.x;
  v = wave_sum(v);
  if ((t & 31) == 0) sh[t >> 5] = v;
  __syncthreads();
  float r = (t < (int)(blockDim.x >> 5)) ? sh[t] : 0.f;
  if ((t >> 5) == 0) { r = wave_sum(r); if (t == 0) sh[0] = r; }
  __syncthreads();
  r = sh[0];
  __syncthreads();
  return r;
}
__device__ inline float block_max(float v, float* sh) {
  int t = threadIdx.x;
  v = wave_max(v);
  if ((t & 31) == 0) sh[t >> 5] = v;
  __syncthreads();
  float r = (t < (int)(blockDim.x >> 5)) ? sh[t] : -1e30f;
  if ((t >> 5) == 0) { r = wave_max(r); if (t == 0) sh[0] = r; }
  __syncthreads();
  r = sh[0];
  __syncthreads();
  return r;
}

// ---------------------------------------------------------------------------
// LayerNorm: f32 rows -> f16 rows (feeds WMMA GEMM directly)
// ---------------------------------------------------------------------------
__global__ __launch_bounds__(256) void layernorm_f16(
    const float* __restrict__ x, const float* __restrict__ g,
    const float* __restrict__ b, h16* __restrict__ y, int cols)
{
  __shared__ float sh[8];
  long long row = blockIdx.x;
  const float* xr = x + row * cols;
  float s = 0.f, s2 = 0.f;
  for (int c = threadIdx.x; c < cols; c += blockDim.x) {
    float v = xr[c]; s += v; s2 += v * v;
  }
  float mean = block_sum(s, sh) / cols;
  float var  = block_sum(s2, sh) / cols - mean * mean;
  float inv  = rsqrtf(var + 1e-5f);
  h16* yr = y + row * cols;
  for (int c = threadIdx.x; c < cols; c += blockDim.x)
    yr[c] = (h16)((xr[c] - mean) * inv * g[c] + b[c]);
}

// ---------------------------------------------------------------------------
// Attention operand prep: one wave32 per (b,h,n) token.
//   Qcat = [q*SCALE || sqrt(clip(cq))]   (K=128)
//   Kcat = [k       || sqrt(clip(ck))]
//   vT/cvT = V^T padded to K=NPAD ; row/col Wasserstein norm terms.
// ---------------------------------------------------------------------------
__global__ __launch_bounds__(256) void attn_prep(
    const float* __restrict__ qkv, const float* __restrict__ cqkv,
    h16* __restrict__ Qcat, h16* __restrict__ Kcat,
    h16* __restrict__ vT, h16* __restrict__ cvT,
    float* __restrict__ rt, float* __restrict__ ct)
{
  int wid = blockIdx.x * (blockDim.x >> 5) + (threadIdx.x >> 5);
  if (wid >= BH * Ntok) return;
  int ln = threadIdx.x & 31;
  int bh = wid / Ntok, n = wid - bh * Ntok;
  int b  = bh / Hh,    h = bh - b * Hh;

  long long rq = ((long long)b * Ntok + n) * QKV3E + h * Dh;
  long long qrow = (long long)wid * 128;
  float qsum = 0.f, ksum = 0.f;
  for (int j = 0; j < 2; ++j) {
    int d = ln + 32 * j;
    float q  = qkv[rq + d] * SCALEQ;
    float k  = qkv[rq + E + d];
    float v  = qkv[rq + 2 * E + d];
    float cq = cqkv[rq + d];
    float ck = cqkv[rq + E + d];
    float cv = cqkv[rq + 2 * E + d];
    float sq = sqrtf(fmaxf(cq, 1e-24f));
    float sk = sqrtf(fmaxf(ck, 1e-24f));
    Qcat[qrow + d]      = (h16)q;
    Qcat[qrow + Dh + d] = (h16)sq;
    Kcat[qrow + d]      = (h16)k;
    Kcat[qrow + Dh + d] = (h16)sk;
    vT [((long long)bh * Dh + d) * NPAD + n] = (h16)v;
    cvT[((long long)bh * Dh + d) * NPAD + n] = (h16)cv;
    qsum += q * q + cq;
    ksum += k * k + ck;
  }
  qsum = wave_sum(qsum);
  ksum = wave_sum(ksum);
  if (ln == 0) { rt[wid] = qsum; ct[wid] = ksum; }
}

// ---------------------------------------------------------------------------
// sigmoid(2G - rt - ct + eps) + rel_pos_bias -> softmax -> attn, attn^2 (f16)
// One 256-thread block per (bh, n) row; writes zero K-padding.
// ---------------------------------------------------------------------------
__global__ __launch_bounds__(256) void attn_softmax(
    const float* __restrict__ G, const float* __restrict__ rt,
    const float* __restrict__ ct, const float* __restrict__ rpb,
    h16* __restrict__ attn, h16* __restrict__ attn2)
{
  __shared__ float sh[8];
  int row = blockIdx.x;                 // bh*Ntok + n
  int bh = row / Ntok, n = row - bh * Ntok;
  int h = bh % Hh;
  const float* grow = G + (long long)bh * Ntok * Ntok + (long long)n * Ntok;
  const float* brow = rpb + ((long long)h * Ntok + n) * Ntok;
  float rtv = rt[row];

  int m = threadIdx.x;
  bool active = (m < Ntok);
  float z = -1e30f;
  if (active) {
    float sarg = 2.f * grow[m] - rtv - ct[bh * Ntok + m] + 1e-24f;
    float s = 1.f / (1.f + __expf(-sarg));
    z = s + brow[m];
  }
  float mx = block_max(z, sh);
  float e = active ? __expf(z - mx) : 0.f;
  float sum = block_sum(e, sh);
  float p = e / sum;
  if (m < NPAD) {
    long long o = (long long)row * NPAD + m;
    attn[o]  = active ? (h16)p : (h16)0.f;
    attn2[o] = active ? (h16)(p * p) : (h16)0.f;
  }
}

// ---------------------------------------------------------------------------
// Small elementwise kernels
// ---------------------------------------------------------------------------
__global__ __launch_bounds__(256) void conv_f32_f16(
    const float* __restrict__ in, h16* __restrict__ out, long long n)
{
  for (long long i = (long long)blockIdx.x * blockDim.x + threadIdx.x; i < n;
       i += (long long)gridDim.x * blockDim.x)
    out[i] = (h16)in[i];
}

// qkv bias: [L, 3E] = concat(q_bias, zeros, v_bias) per layer
__global__ __launch_bounds__(256) void build_qkv_bias(
    const float* __restrict__ qb, const float* __restrict__ vb,
    float* __restrict__ out)
{
  long long i = (long long)blockIdx.x * blockDim.x + threadIdx.x;
  if (i >= (long long)Lyr * QKV3E) return;
  int l = (int)(i / QKV3E), j = (int)(i - (long long)l * QKV3E);
  float v = 0.f;
  if (j < E)            v = qb[l * E + j];
  else if (j >= 2 * E)  v = vb[l * E + (j - 2 * E)];
  out[i] = v;
}

// im2col for 16x16/stride-16 patch embed: A[B*196, 768] f16
__global__ __launch_bounds__(256) void im2col_patches(
    const float* __restrict__ x, h16* __restrict__ out)
{
  long long i = (long long)blockIdx.x * blockDim.x + threadIdx.x;
  long long tot = (long long)PE_ROWS * E;
  if (i >= tot) return;
  int k = (int)(i % E);
  int row = (int)(i / E);
  int b = row / NPATCH, p = row - b * NPATCH;
  int pi = p / 14, pj = p - pi * 14;
  int c = k >> 8, r = k & 255, py = r >> 4, px = r & 15;
  long long src = (((long long)b * 3 + c) * 224 + (pi * 16 + py)) * 224 +
                  (pj * 16 + px);
  out[i] = (h16)x[src];
}

// tokens: row0 = cls, rows 1..196 = patch-embed rows
__global__ __launch_bounds__(256) void assemble_tokens(
    const float* __restrict__ pe, const float* __restrict__ cls,
    float* __restrict__ xm)
{
  long long i = (long long)blockIdx.x * blockDim.x + threadIdx.x;
  long long tot = (long long)BN * E;
  if (i >= tot) return;
  int e = (int)(i % E);
  int rn = (int)(i / E);
  int n = rn % Ntok, b = rn / Ntok;
  xm[i] = (n == 0) ? cls[e] : pe[((long long)b * NPATCH + (n - 1)) * E + e];
}

// ctx [BH, Ntok, 64] f32 -> [B*Ntok, 768] f16
__global__ __launch_bounds__(256) void reshape_ctx(
    const float* __restrict__ ctx, h16* __restrict__ out)
{
  long long i = (long long)blockIdx.x * blockDim.x + threadIdx.x;
  long long tot = (long long)BN * E;
  if (i >= tot) return;
  int e = (int)(i % E);
  int rn = (int)(i / E);
  int n = rn % Ntok, b = rn / Ntok;
  int h = e >> 6, d = e & 63;
  out[i] = (h16)ctx[(((long long)(b * Hh + h)) * Ntok + n) * Dh + d];
}

// x += gamma[col] * delta
__global__ __launch_bounds__(256) void residual_gamma(
    float* __restrict__ x, const float* __restrict__ delta,
    const float* __restrict__ gamma, long long n)
{
  long long i = (long long)blockIdx.x * blockDim.x + threadIdx.x;
  if (i >= n) return;
  x[i] += gamma[i % E] * delta[i];
}

// mean over patch tokens (n = 1..196)
__global__ __launch_bounds__(256) void pool_tokens(
    const float* __restrict__ xm, float* __restrict__ t)
{
  int i = blockIdx.x * blockDim.x + threadIdx.x;
  if (i >= Bsz * E) return;
  int b = i / E, e = i - b * E;
  float s = 0.f;
  for (int n = 1; n < Ntok; ++n) s += xm[((long long)b * Ntok + n) * E + e];
  t[i] = s / (float)NPATCH;
}

// ---------------------------------------------------------------------------
// Host orchestration
// ---------------------------------------------------------------------------
extern "C" void kernel_launch(void* const* d_in, const int* in_sizes, int n_in,
                              void* d_out, int out_size, void* d_ws,
                              size_t ws_size, hipStream_t stream) {
  (void)in_sizes; (void)n_in; (void)out_size; (void)ws_size;

  const float* X        = (const float*)d_in[0];
  const float* rpb      = (const float*)d_in[1];
  const float* patch_w  = (const float*)d_in[2];
  const float* patch_b  = (const float*)d_in[3];
  const float* cpatch_w = (const float*)d_in[4];
  const float* cpatch_b = (const float*)d_in[5];
  const float* cls_tok  = (const float*)d_in[6];
  const float* ccls_tok = (const float*)d_in[7];
  const float* n1g      = (const float*)d_in[8];
  const float* n1b      = (const float*)d_in[9];
  const float* qkv_w    = (const float*)d_in[10];
  const float* q_bias   = (const float*)d_in[11];
  const float* v_bias   = (const float*)d_in[12];
  const float* cq_bias  = (const float*)d_in[13];
  const float* cv_bias  = (const float*)d_in[14];
  const float* proj_w   = (const float*)d_in[15];
  const float* proj_b   = (const float*)d_in[16];
  const float* cproj_w  = (const float*)d_in[17];
  const float* cproj_b  = (const float*)d_in[18];
  const float* gamma1   = (const float*)d_in[19];
  const float* gamma2   = (const float*)d_in[20];
  const float* n2g      = (const float*)d_in[21];
  const float* n2b      = (const float*)d_in[22];
  const float* fc1_w    = (const float*)d_in[23];
  const float* fc1_b    = (const float*)d_in[24];
  const float* fc2_w    = (const float*)d_in[25];
  const float* fc2_b    = (const float*)d_in[26];
  const float* fng      = (const float*)d_in[27];
  const float* fnb      = (const float*)d_in[28];
  const float* head_w   = (const float*)d_in[29];
  const float* head_b   = (const float*)d_in[30];
  float* out = (float*)d_out;

  // ---- bump allocator over d_ws (deterministic layout every call) ----
  char* wsp = (char*)d_ws;
  size_t off = 0;
  auto alloc = [&](size_t bytes) -> void* {
    void* p = wsp + off;
    off = (off + bytes + 255) & ~(size_t)255;
    return p;
  };
  auto allocF = [&](size_t n) { return (float*)alloc(n * sizeof(float)); };
  auto allocH = [&](size_t n) { return (h16*)alloc(n * sizeof(h16)); };

  // f16 weights
  h16* w16_qkv   = allocH((size_t)Lyr * QKV3E * E);
  h16* w16_proj  = allocH((size_t)Lyr * E * E);
  h16* w16_cproj = allocH((size_t)Lyr * E * E);
  h16* w16_fc1   = allocH((size_t)Lyr * MLPD * E);
  h16* w16_fc2   = allocH((size_t)Lyr * E * MLPD);
  h16* w16_patch = allocH((size_t)E * E);
  h16* w16_cpatch= allocH((size_t)E * E);
  h16* w16_head  = allocH((size_t)NCLS * E);
  float* qkvb_m  = allocF((size_t)Lyr * QKV3E);
  float* qkvb_c  = allocF((size_t)Lyr * QKV3E);

  // activations
  h16*   A_pe   = allocH((size_t)PE_ROWS * E);
  float* pe_m   = allocF((size_t)PE_ROWS * E);
  float* pe_c   = allocF((size_t)PE_ROWS * E);
  float* xm     = allocF((size_t)BN * E);
  float* xc     = allocF((size_t)BN * E);
  h16*   lnm16  = allocH((size_t)BN * E);
  h16*   lnc16  = allocH((size_t)BN * E);
  float* qkv_m  = allocF((size_t)BN * QKV3E);
  float* qkv_c  = allocF((size_t)BN * QKV3E);
  h16*   Qcat   = allocH((size_t)BH * Ntok * 128);
  h16*   Kcat   = allocH((size_t)BH * Ntok * 128);
  float* rt     = allocF((size_t)BH * Ntok);
  float* ct     = allocF((size_t)BH * Ntok);
  h16*   vT16   = allocH((size_t)BH * Dh * NPAD);
  h16*   cvT16  = allocH((size_t)BH * Dh * NPAD);
  float* Gm     = allocF((size_t)BH * Ntok * Ntok);
  h16*   attn16 = allocH((size_t)BH * Ntok * NPAD);
  h16*   attn2  = allocH((size_t)BH * Ntok * NPAD);
  float* ctx    = allocF((size_t)BH * Ntok * Dh);
  float* cctx   = allocF((size_t)BH * Ntok * Dh);
  h16*   ctxh16 = allocH((size_t)BN * E);
  h16*   cctxh16= allocH((size_t)BN * E);
  float* mh     = allocF((size_t)BN * E);
  float* ch     = allocF((size_t)BN * E);
  float* h1     = allocF((size_t)BN * MLPD);
  h16*   h1_16  = allocH((size_t)BN * MLPD);
  float* tpool  = allocF((size_t)Bsz * E);
  h16*   t16    = allocH((size_t)Bsz * E);

  auto cvt = [&](const float* src, h16* dst, long long n) {
    int g = (int)((n + 255) / 256); if (g > 4096) g = 4096;
    conv_f32_f16<<<g, 256, 0, stream>>>(src, dst, n);
  };
  auto gemm = [&](const h16* A, long long sA, int lda, const h16* W,
                  long long sW, int ldw, float* C, long long sC, int ldc,
                  const float* bias, h16* C16, int ldc16, int M, int N, int K,
                  int act, int batch) {
    dim3 g((N + TN - 1) / TN, (M + TM - 1) / TM, batch);
    wmma_gemm_f16<<<g, 256, SMEMB, stream>>>(A, sA, lda, W, sW, ldw, C, sC,
                                             ldc, bias, C16, ldc16, M, N, K,
                                             act);
  };

  // ---- weight conversion (f32 -> f16) ----
  cvt(qkv_w,   w16_qkv,   (long long)Lyr * QKV3E * E);
  cvt(proj_w,  w16_proj,  (long long)Lyr * E * E);
  cvt(cproj_w, w16_cproj, (long long)Lyr * E * E);
  cvt(fc1_w,   w16_fc1,   (long long)Lyr * MLPD * E);
  cvt(fc2_w,   w16_fc2,   (long long)Lyr * E * MLPD);
  cvt(patch_w, w16_patch, (long long)E * E);
  cvt(cpatch_w,w16_cpatch,(long long)E * E);
  cvt(head_w,  w16_head,  (long long)NCLS * E);
  {
    int g = (Lyr * QKV3E + 255) / 256;
    build_qkv_bias<<<g, 256, 0, stream>>>(q_bias, v_bias, qkvb_m);
    build_qkv_bias<<<g, 256, 0, stream>>>(cq_bias, cv_bias, qkvb_c);
  }

  // ---- patch embed (im2col + WMMA GEMM) ----
  {
    long long tot = (long long)PE_ROWS * E;
    im2col_patches<<<(int)((tot + 255) / 256), 256, 0, stream>>>(X, A_pe);
    gemm(A_pe, 0, E, w16_patch, 0, E, pe_m, 0, E, patch_b, nullptr, 0,
         PE_ROWS, E, E, 0, 1);
    gemm(A_pe, 0, E, w16_cpatch, 0, E, pe_c, 0, E, cpatch_b, nullptr, 0,
         PE_ROWS, E, E, 0, 1);
    int g = (int)(((long long)BN * E + 255) / 256);
    assemble_tokens<<<g, 256, 0, stream>>>(pe_m, cls_tok, xm);
    assemble_tokens<<<g, 256, 0, stream>>>(pe_c, ccls_tok, xc);
  }

  // zero V^T pads once (prep only writes n < Ntok; pads stay zero)
  hipMemsetAsync(vT16,  0, (size_t)BH * Dh * NPAD * sizeof(h16), stream);
  hipMemsetAsync(cvT16, 0, (size_t)BH * Dh * NPAD * sizeof(h16), stream);

  const int gEw = (int)(((long long)BN * E + 255) / 256);

  for (int i = 0; i < Lyr; ++i) {
    // norm1 (shared params for both streams)
    layernorm_f16<<<BN, 256, 0, stream>>>(xm, n1g + i * E, n1b + i * E, lnm16, E);
    layernorm_f16<<<BN, 256, 0, stream>>>(xc, n1g + i * E, n1b + i * E, lnc16, E);

    // QKV GEMMs (cov path bug-faithfully reuses qkv_w; epilogue elu+1)
    const h16* wq = w16_qkv + (size_t)i * QKV3E * E;
    gemm(lnm16, 0, E, wq, 0, E, qkv_m, 0, QKV3E, qkvb_m + i * QKV3E,
         nullptr, 0, BN, QKV3E, E, 0, 1);
    gemm(lnc16, 0, E, wq, 0, E, qkv_c, 0, QKV3E, qkvb_c + i * QKV3E,
         nullptr, 0, BN, QKV3E, E, 2, 1);

    // attention operand prep
    attn_prep<<<(BH * Ntok + 7) / 8, 256, 0, stream>>>(
        qkv_m, qkv_c, Qcat, Kcat, vT16, cvT16, rt, ct);

    // Wasserstein score core: G = Qcat . Kcat^T  (batched, K=128)
    gemm(Qcat, (long long)Ntok * 128, 128, Kcat, (long long)Ntok * 128, 128,
         Gm, (long long)Ntok * Ntok, Ntok, nullptr, nullptr, 0,
         Ntok, Ntok, 128, 0, BH);

    // sigmoid + rel_pos_bias + softmax -> attn / attn^2 (f16, K-padded)
    attn_softmax<<<BH * Ntok, 256, 0, stream>>>(Gm, rt, ct, rpb, attn16, attn2);

    // context: attn @ V ; attn^2 @ covV  (batched, K=NPAD)
    gemm(attn16, (long long)Ntok * NPAD, NPAD, vT16, (long long)Dh * NPAD,
         NPAD, ctx, (long long)Ntok * Dh, Dh, nullptr, nullptr, 0,
         Ntok, Dh, NPAD, 0, BH);
    gemm(attn2, (long long)Ntok * NPAD, NPAD, cvT16, (long long)Dh * NPAD,
         NPAD, cctx, (long long)Ntok * Dh, Dh, nullptr, nullptr, 0,
         Ntok, Dh, NPAD, 0, BH);

    reshape_ctx<<<gEw, 256, 0, stream>>>(ctx, ctxh16);
    reshape_ctx<<<gEw, 256, 0, stream>>>(cctx, cctxh16);

    // output projections + gamma1 residuals
    gemm(ctxh16, 0, E, w16_proj + (size_t)i * E * E, 0, E, mh, 0, E,
         proj_b + i * E, nullptr, 0, BN, E, E, 0, 1);
    gemm(cctxh16, 0, E, w16_cproj + (size_t)i * E * E, 0, E, ch, 0, E,
         cproj_b + i * E, nullptr, 0, BN, E, E, 0, 1);
    residual_gamma<<<gEw, 256, 0, stream>>>(xm, mh, gamma1 + i * E, (long long)BN * E);
    residual_gamma<<<gEw, 256, 0, stream>>>(xc, ch, gamma1 + i * E, (long long)BN * E);

    // MLP (shared weights across streams), gamma2 residuals
    const h16* wf1 = w16_fc1 + (size_t)i * MLPD * E;
    const h16* wf2 = w16_fc2 + (size_t)i * E * MLPD;
    // mean stream
    layernorm_f16<<<BN, 256, 0, stream>>>(xm, n2g + i * E, n2b + i * E, lnm16, E);
    gemm(lnm16, 0, E, wf1, 0, E, h1, 0, MLPD, fc1_b + i * MLPD, h1_16, MLPD,
         BN, MLPD, E, 1, 1);                       // GELU epilogue, f16 mirror
    gemm(h1_16, 0, MLPD, wf2, 0, MLPD, mh, 0, E, fc2_b + i * E, nullptr, 0,
         BN, E, MLPD, 0, 1);
    residual_gamma<<<gEw, 256, 0, stream>>>(xm, mh, gamma2 + i * E, (long long)BN * E);
    // cov stream
    layernorm_f16<<<BN, 256, 0, stream>>>(xc, n2g + i * E, n2b + i * E, lnc16, E);
    gemm(lnc16, 0, E, wf1, 0, E, h1, 0, MLPD, fc1_b + i * MLPD, h1_16, MLPD,
         BN, MLPD, E, 1, 1);
    gemm(h1_16, 0, MLPD, wf2, 0, MLPD, ch, 0, E, fc2_b + i * E, nullptr, 0,
         BN, E, MLPD, 0, 1);
    residual_gamma<<<gEw, 256, 0, stream>>>(xc, ch, gamma2 + i * E, (long long)BN * E);
  }

  // ---- head: mean-pool patch tokens, fc_norm, classifier GEMM ----
  pool_tokens<<<(Bsz * E + 255) / 256, 256, 0, stream>>>(xm, tpool);
  layernorm_f16<<<Bsz, 256, 0, stream>>>(tpool, fng, fnb, t16, E);
  gemm(t16, 0, E, w16_head, 0, E, out, 0, NCLS, head_b, nullptr, 0,
       Bsz, NCLS, E, 0, 1);
}
